// MSA2Pair_76020921139678
// MI455X (gfx1250) — matrix operand order repaired and checked by reading, not compiled, not run on hardware
//
#include <hip/hip_runtime.h>
#include <hip/hip_bf16.h>

// ---------------- types & helpers ----------------
typedef __attribute__((ext_vector_type(8)))  float  v8f;
typedef __attribute__((ext_vector_type(8)))  __bf16 v8bf;
typedef __attribute__((ext_vector_type(16))) __bf16 v16bf;

__device__ __forceinline__ unsigned short f2bf(float f) {
  unsigned u = __builtin_bit_cast(unsigned, f);
  unsigned r = u + 0x7FFFu + ((u >> 16) & 1u);   // round-to-nearest-even
  return (unsigned short)(r >> 16);
}

__device__ __forceinline__ float wred(float v) {
#pragma unroll
  for (int m = 16; m > 0; m >>= 1) v += __shfl_xor(v, m, 32);
  return v;
}

// load 16 bf16 as two contiguous 8-element (16B) chunks
__device__ __forceinline__ v16bf load16(const unsigned short* p0,
                                        const unsigned short* p1) {
  v8bf lo = *(const v8bf*)p0;
  v8bf hi = *(const v8bf*)p1;
  v16bf r;
#pragma unroll
  for (int i = 0; i < 8; ++i) { r[i] = lo[i]; r[i + 8] = hi[i]; }
  return r;
}

__device__ __forceinline__ v8f wmma_bf16(v16bf a, v16bf b, v8f c) {
  return __builtin_amdgcn_wmma_f32_16x16x32_bf16(
      false, a, false, b, (short)0, c, false, false);
}

// ---------------- constants ----------------
#define NN 256
#define LL 256
#define FF 64
#define PP 32
#define CC 128
#define FD 384
#define EPS 1e-5f

// ws offsets (bytes), all 4KB-aligned; total ~48.7 MB
#define OFF_XD    0ull                        // f32  x_down [n][l][p]      8 MB
#define OFF_XDT   8388608ull                  // bf16 xdT [(l*32+p)][n]     4 MB
#define OFF_FDT   12582912ull                 // bf16 fdT [(l*32+q)][n]     4 MB
#define OFF_FEAT  16777216ull                 // f32  feat [l][64]          64 KB
#define OFF_PAIR  16842752ull                 // f32  pair [i][j][c]        32 MB
#define OFF_W2T   50397184ull                 // bf16 w2T  [c][1024]
#define OFF_WF1T  50659328ull                 // bf16 wf1T [n][384]
#define OFF_WF2T  50954240ull                 // bf16 wf2T [c][384]

// ---------------- K0: weight conversion ----------------
__global__ void k_wconv(const float* __restrict__ W2,
                        const float* __restrict__ Wf1,
                        const float* __restrict__ Wf2,
                        unsigned short* __restrict__ w2T,
                        unsigned short* __restrict__ wf1T,
                        unsigned short* __restrict__ wf2T) {
  int t = blockIdx.x * 256 + threadIdx.x;
  if (t < 1024 * 128) { int k = t >> 7, c = t & 127; w2T[c * 1024 + k] = f2bf(W2[t]); }
  if (t < 384 * 384)  { int k = t / 384, n = t % 384; wf1T[n * 384 + k] = f2bf(Wf1[t]); }
  if (t < 384 * 128)  { int k = t >> 7, c = t & 127; wf2T[c * 384 + k] = f2bf(Wf2[t]); }
}

// ---------------- K1: x_down = ln(ln(msa) @ Wp + bp) ----------------
// one wave per (n,l) row; 8 rows per 256-thread block
__global__ void k_xdown(const float* __restrict__ msa,
                        const float* __restrict__ g1, const float* __restrict__ b1,
                        const float* __restrict__ Wp, const float* __restrict__ bp,
                        const float* __restrict__ gd, const float* __restrict__ bd,
                        float* __restrict__ xd, unsigned short* __restrict__ xdT) {
  __shared__ float xn[8][FF];
  int wave = threadIdx.x >> 5, lane = threadIdx.x & 31;
  int row = blockIdx.x * 8 + wave;           // row = n*256 + l
  int n = row >> 8, l = row & 255;
  const float* m = msa + (size_t)row * FF;
  float x0 = m[lane], x1 = m[lane + 32];
  float s  = wred(x0 + x1) * (1.f / 64.f);
  float ss = wred(x0 * x0 + x1 * x1) * (1.f / 64.f);
  float rs = rsqrtf(ss - s * s + EPS);
  xn[wave][lane]      = (x0 - s) * rs * g1[lane]      + b1[lane];
  xn[wave][lane + 32] = (x1 - s) * rs * g1[lane + 32] + b1[lane + 32];
  __syncthreads();
  float acc = bp[lane];
#pragma unroll 8
  for (int f = 0; f < FF; ++f) acc += xn[wave][f] * Wp[f * PP + lane];
  float m2 = wred(acc) * (1.f / 32.f);
  float v2 = wred(acc * acc) * (1.f / 32.f) - m2 * m2;
  float o  = (acc - m2) * rsqrtf(v2 + EPS) * gd[lane] + bd[lane];
  xd[(size_t)row * PP + lane] = o;
  xdT[(size_t)(l * PP + lane) * NN + n] = f2bf(o);
}

// ---------------- K2: attention over sequences, feat_1d, f1, feat ----------------
// one block (256 threads) per position l; thread t handles sequence n=t
__global__ void k_attn(const float* __restrict__ xd,
                       const float* __restrict__ Wq, const float* __restrict__ bq,
                       const float* __restrict__ Wk, const float* __restrict__ bk,
                       unsigned short* __restrict__ fdT, float* __restrict__ feat) {
  __shared__ float tar[PP], qv[PP], vv[PP], red[256], f1s[PP];
  __shared__ float c0s;
  int l = blockIdx.x, t = threadIdx.x;
  if (t < PP) { tar[t] = xd[(size_t)l * PP + t]; f1s[t] = 0.f; }  // n=0 row
  __syncthreads();
  if (t < PP) {
    float a = bq[t];
#pragma unroll 8
    for (int p = 0; p < PP; ++p) a += tar[p] * Wq[p * PP + t];
    qv[t] = a * 0.17677669529663687f;   // 1/sqrt(32)
  }
  __syncthreads();
  if (t < PP) {
    float a = 0.f;
#pragma unroll 8
    for (int p = 0; p < PP; ++p) a += Wk[t * PP + p] * qv[p];
    vv[t] = a;
  }
  if (t == 0) {
    float c = 0.f;
    for (int p = 0; p < PP; ++p) c += qv[p] * bk[p];
    c0s = c;
  }
  __syncthreads();
  // logits[n] = x_down[n,l,:] . vv + c0
  float xr[PP];
  const float* xp = xd + (size_t)(t * LL + l) * PP;
  float lg = c0s;
#pragma unroll
  for (int p = 0; p < PP; ++p) { xr[p] = xp[p]; lg += xr[p] * vv[p]; }
  red[t] = lg; __syncthreads();
  for (int sft = 128; sft > 0; sft >>= 1) {
    if (t < sft) red[t] = fmaxf(red[t], red[t + sft]);
    __syncthreads();
  }
  float mx = red[0]; __syncthreads();
  float e = __expf(lg - mx);
  red[t] = e; __syncthreads();
  for (int sft = 128; sft > 0; sft >>= 1) {
    if (t < sft) red[t] += red[t + sft];
    __syncthreads();
  }
  float w = e / red[0]; __syncthreads();
#pragma unroll
  for (int p = 0; p < PP; ++p) {
    float fv = w * xr[p];
    fdT[(size_t)(l * PP + p) * NN + t] = f2bf(fv);
    atomicAdd(&f1s[p], fv);
  }
  __syncthreads();
  if (t < PP) { feat[l * 64 + t] = f1s[t]; feat[l * 64 + 32 + t] = tar[t]; }
}

// ---------------- K4: fused op -> ln(1024) -> @W2 + b2 ----------------
// 256 threads (8 waves). Each wave: two (i,j) 32x32 blocks via 2x2 WMMA tiles,
// K=256 (8 bf16 k-steps). Then 16x1024 @ 1024x128 WMMA out of LDS.
__global__ void __launch_bounds__(256)
k_op_pair(const unsigned short* __restrict__ xdT,
          const unsigned short* __restrict__ fdT,
          const float* __restrict__ g2d, const float* __restrict__ b2d,
          const unsigned short* __restrict__ w2T, const float* __restrict__ b2,
          float* __restrict__ pair) {
  __shared__ __align__(16) unsigned short Ablk[16][1024];   // 32 KB
  int wave = threadIdx.x >> 5, lane = threadIdx.x & 31;
  int l16 = lane & 15, hi = lane >> 4;
  int kA0 = hi ? 16 : 0;    // A-operand: elems[0:16]=K[kA0..kA0+16)
  int kB0 = hi ? 16 : 0;    // B-operand: elems[0:16]=K[kB0..kB0+16)
  int wgBase = blockIdx.x * 16;

  for (int blk = 0; blk < 2; ++blk) {
    int bidx = wgBase + wave * 2 + blk;      // = i*256 + j
    int i = bidx >> 8, j = bidx & 255;
    const unsigned short* a0p = xdT + (size_t)(i * PP + l16) * NN;
    const unsigned short* a1p = xdT + (size_t)(i * PP + 16 + l16) * NN;
    const unsigned short* b0p = fdT + (size_t)(j * PP + l16) * NN;
    const unsigned short* b1p = fdT + (size_t)(j * PP + 16 + l16) * NN;
    v8f c00 = {}, c01 = {}, c10 = {}, c11 = {};
#pragma unroll
    for (int kk = 0; kk < 8; ++kk) {
      int kb = kk * 32;
      v16bf a0 = load16(a0p + kb + kA0, a0p + kb + kA0 + 8);
      v16bf a1 = load16(a1p + kb + kA0, a1p + kb + kA0 + 8);
      v16bf b0 = load16(b0p + kb + kB0, b0p + kb + kB0 + 8);
      v16bf b1 = load16(b1p + kb + kB0, b1p + kb + kB0 + 8);
      c00 = wmma_bf16(a0, b0, c00);
      c01 = wmma_bf16(a0, b1, c01);
      c10 = wmma_bf16(a1, b0, c10);
      c11 = wmma_bf16(a1, b1, c11);
    }
    // layernorm over the 1024 values of this (i,j) block (one wave holds all)
    float s = 0.f, ss = 0.f;
#pragma unroll
    for (int r = 0; r < 8; ++r) {
      s  += c00[r] + c01[r] + c10[r] + c11[r];
      ss += c00[r] * c00[r] + c01[r] * c01[r] + c10[r] * c10[r] + c11[r] * c11[r];
    }
    s  = wred(s)  * (1.f / 1024.f);
    ss = wred(ss) * (1.f / 1024.f);
    float rs = rsqrtf(ss - s * s + EPS);
    int rowL = wave * 2 + blk;
#pragma unroll
    for (int r = 0; r < 8; ++r) {
      int p0 = r + hi * 8, q0 = l16, pq;
      pq = p0 * 32 + q0;             Ablk[rowL][pq] = f2bf((c00[r] - s) * rs * g2d[pq] + b2d[pq]);
      pq = p0 * 32 + q0 + 16;        Ablk[rowL][pq] = f2bf((c01[r] - s) * rs * g2d[pq] + b2d[pq]);
      pq = (p0 + 16) * 32 + q0;      Ablk[rowL][pq] = f2bf((c10[r] - s) * rs * g2d[pq] + b2d[pq]);
      pq = (p0 + 16) * 32 + q0 + 16; Ablk[rowL][pq] = f2bf((c11[r] - s) * rs * g2d[pq] + b2d[pq]);
    }
  }
  __syncthreads();
  // phase 2: 16 (blocks) x 1024 @ 1024 x 128; one 16-wide N tile per wave
  {
    int c = wave * 16 + l16;
    const unsigned short* bcol = w2T + (size_t)c * 1024;
    v8f acc = {};
    for (int kk = 0; kk < 32; ++kk) {
      int kb = kk * 32;
      const unsigned short* ar = &Ablk[l16][kb + kA0];
      v16bf a = load16(ar, ar + 8);
      v16bf b = load16(bcol + kb + kB0, bcol + kb + kB0 + 8);
      acc = wmma_bf16(a, b, acc);
    }
    float bb = b2[c];
#pragma unroll
    for (int r = 0; r < 8; ++r) {
      int M = r + hi * 8;
      pair[(size_t)(wgBase + M) * CC + c] = acc[r] + bb;
    }
  }
}

// ---------------- K5: fused cat -> ln -> relu(@Wf1) -> @Wf2 + residual ----------------
// 256 threads handle 16 output rows (i,j); one wave builds 2 cat rows.
__global__ void __launch_bounds__(256)
k_ffn(const float* __restrict__ pair_orig, const float* __restrict__ pair,
      const float* __restrict__ feat,
      const float* __restrict__ go, const float* __restrict__ bo,
      const float* __restrict__ gn, const float* __restrict__ bn,
      const float* __restrict__ gu, const float* __restrict__ bu,
      const unsigned short* __restrict__ wf1T, const float* __restrict__ bf1,
      const unsigned short* __restrict__ wf2T, const float* __restrict__ bf2,
      float* __restrict__ out) {
  __shared__ __align__(16) unsigned short Acat[16][FD];   // 12 KB
  __shared__ __align__(16) unsigned short H1[16][FD];     // 12 KB
  __shared__ float Pon[16][CC];                           //  8 KB
  int wave = threadIdx.x >> 5, lane = threadIdx.x & 31;
  int l16 = lane & 15, hi = lane >> 4;
  int kA0 = hi ? 16 : 0, kB0 = hi ? 16 : 0;
  int wgBase = blockIdx.x * 16;

  for (int rr = 0; rr < 2; ++rr) {
    int rowL = wave * 2 + rr;
    int r = wgBase + rowL;
    int i = r >> 8, j = r & 255;
    const float* pop = pair_orig + (size_t)r * CC;
    const float* prp = pair + (size_t)r * CC;
    float po[4], pr[4], fi[2], fj[2];
    float s1 = 0.f, q1 = 0.f;
#pragma unroll
    for (int u = 0; u < 4; ++u) { po[u] = pop[lane + 32 * u]; s1 += po[u]; q1 += po[u] * po[u]; }
    s1 = wred(s1) * (1.f / 128.f); q1 = wred(q1) * (1.f / 128.f);
    float rs1 = rsqrtf(q1 - s1 * s1 + EPS);
#pragma unroll
    for (int u = 0; u < 4; ++u) {
      int idx = lane + 32 * u;
      po[u] = (po[u] - s1) * rs1 * go[idx] + bo[idx];
      Pon[rowL][idx] = po[u];
    }
    float s3 = 0.f, q3 = 0.f;
#pragma unroll
    for (int u = 0; u < 4; ++u) { pr[u] = prp[lane + 32 * u]; s3 += pr[u]; q3 += pr[u] * pr[u]; }
    s3 = wred(s3) * (1.f / 128.f); q3 = wred(q3) * (1.f / 128.f);
    float rs3 = rsqrtf(q3 - s3 * s3 + EPS);
#pragma unroll
    for (int u = 0; u < 4; ++u) {
      int idx = lane + 32 * u;
      pr[u] = (pr[u] - s3) * rs3 * gn[idx] + bn[idx];
    }
#pragma unroll
    for (int u = 0; u < 2; ++u) {
      fi[u] = feat[i * 64 + lane + 32 * u];
      fj[u] = feat[j * 64 + lane + 32 * u];
    }
    float s = 0.f, q = 0.f;
#pragma unroll
    for (int u = 0; u < 4; ++u) { s += po[u] + pr[u]; q += po[u] * po[u] + pr[u] * pr[u]; }
#pragma unroll
    for (int u = 0; u < 2; ++u) { s += fi[u] + fj[u]; q += fi[u] * fi[u] + fj[u] * fj[u]; }
    s = wred(s) * (1.f / 384.f); q = wred(q) * (1.f / 384.f);
    float rs = rsqrtf(q - s * s + EPS);
#pragma unroll
    for (int u = 0; u < 4; ++u) {
      int cp = lane + 32 * u;
      Acat[rowL][cp]       = f2bf((po[u] - s) * rs * gu[cp]       + bu[cp]);
      Acat[rowL][cp + 128] = f2bf((pr[u] - s) * rs * gu[cp + 128] + bu[cp + 128]);
    }
#pragma unroll
    for (int u = 0; u < 2; ++u) {
      int cp = 256 + lane + 32 * u;
      Acat[rowL][cp]      = f2bf((fi[u] - s) * rs * gu[cp]      + bu[cp]);
      Acat[rowL][cp + 64] = f2bf((fj[u] - s) * rs * gu[cp + 64] + bu[cp + 64]);
    }
  }
  __syncthreads();
  // h1 = relu(Acat @ Wf1 + bf1): 16 x 384 @ 384 x 384, 3 N-tiles per wave
  for (int st = 0; st < 3; ++st) {
    int n = (wave * 3 + st) * 16 + l16;
    const unsigned short* bcol = wf1T + (size_t)n * FD;
    v8f acc = {};
#pragma unroll
    for (int kk = 0; kk < 12; ++kk) {
      int kb = kk * 32;
      const unsigned short* ar = &Acat[l16][kb + kA0];
      v16bf a = load16(ar, ar + 8);
      v16bf b = load16(bcol + kb + kB0, bcol + kb + kB0 + 8);
      acc = wmma_bf16(a, b, acc);
    }
    float bb = bf1[n];
#pragma unroll
    for (int rj = 0; rj < 8; ++rj) {
      int M = rj + hi * 8;
      float h = acc[rj] + bb;
      H1[M][n] = f2bf(h > 0.f ? h : 0.f);
    }
  }
  __syncthreads();
  // out = Pon + H1 @ Wf2 + bf2: 16 x 384 @ 384 x 128, one N-tile per wave
  {
    int c = wave * 16 + l16;
    const unsigned short* bcol = wf2T + (size_t)c * FD;
    v8f acc = {};
#pragma unroll
    for (int kk = 0; kk < 12; ++kk) {
      int kb = kk * 32;
      const unsigned short* ar = &H1[l16][kb + kA0];
      v16bf a = load16(ar, ar + 8);
      v16bf b = load16(bcol + kb + kB0, bcol + kb + kB0 + 8);
      acc = wmma_bf16(a, b, acc);
    }
    float bb = bf2[c];
#pragma unroll
    for (int rj = 0; rj < 8; ++rj) {
      int M = rj + hi * 8;
      out[(size_t)(wgBase + M) * CC + c] = Pon[M][c] + acc[rj] + bb;
    }
  }
}

// ---------------- launch ----------------
extern "C" void kernel_launch(void* const* d_in, const int* in_sizes, int n_in,
                              void* d_out, int out_size, void* d_ws, size_t ws_size,
                              hipStream_t stream) {
  (void)in_sizes; (void)n_in; (void)out_size; (void)ws_size;
  const float* msa       = (const float*)d_in[0];
  const float* pair_orig = (const float*)d_in[1];
  const float* g1  = (const float*)d_in[2];
  const float* b1  = (const float*)d_in[3];
  const float* Wp  = (const float*)d_in[4];
  const float* bp  = (const float*)d_in[5];
  const float* gd  = (const float*)d_in[6];
  const float* bd  = (const float*)d_in[7];
  const float* Wq  = (const float*)d_in[8];
  const float* bq  = (const float*)d_in[9];
  const float* Wk  = (const float*)d_in[10];
  const float* bk  = (const float*)d_in[11];
  const float* g2d = (const float*)d_in[12];
  const float* b2d = (const float*)d_in[13];
  const float* W2  = (const float*)d_in[14];
  const float* b2  = (const float*)d_in[15];
  const float* go  = (const float*)d_in[16];
  const float* bo  = (const float*)d_in[17];
  const float* gn  = (const float*)d_in[18];
  const float* bn  = (const float*)d_in[19];
  const float* gu  = (const float*)d_in[20];
  const float* bu  = (const float*)d_in[21];
  const float* Wf1 = (const float*)d_in[22];
  const float* bf1 = (const float*)d_in[23];
  const float* Wf2 = (const float*)d_in[24];
  const float* bf2 = (const float*)d_in[25];

  char* ws = (char*)d_ws;
  float*          xd    = (float*)(ws + OFF_XD);
  unsigned short* xdT   = (unsigned short*)(ws + OFF_XDT);
  unsigned short* fdT   = (unsigned short*)(ws + OFF_FDT);
  float*          feat  = (float*)(ws + OFF_FEAT);
  float*          pairb = (float*)(ws + OFF_PAIR);
  unsigned short* w2T   = (unsigned short*)(ws + OFF_W2T);
  unsigned short* wf1T  = (unsigned short*)(ws + OFF_WF1T);
  unsigned short* wf2T  = (unsigned short*)(ws + OFF_WF2T);
  float* out = (float*)d_out;

  k_wconv<<<576, 256, 0, stream>>>(W2, Wf1, Wf2, w2T, wf1T, wf2T);
  k_xdown<<<8192, 256, 0, stream>>>(msa, g1, b1, Wp, bp, gd, bd, xd, xdT);
  k_attn<<<256, 256, 0, stream>>>(xd, Wq, bq, Wk, bk, fdT, feat);
  k_op_pair<<<4096, 256, 0, stream>>>(xdT, fdT, g2d, b2d, w2T, b2, pairb);
  k_ffn<<<4096, 256, 0, stream>>>(pair_orig, pairb, feat, go, bo, gn, bn,
                                  gu, bu, wf1T, bf1, wf2T, bf2, out);
}
